// RNNDecoder_44702019616928
// MI455X (gfx1250) — matrix operand order repaired and checked by reading, not compiled, not run on hardware
//
#include <hip/hip_runtime.h>
#include <hip/hip_bf16.h>

// RNN decoder: embed -> 2-layer GRU -> vocab projection, for MI455X (gfx1250).
// All GEMMs use v_wmma_f32_16x16x32_bf16 (bf16 operands, f32 accum).
// Wave tile 32x64; ping-pong (unroll-2) K pipeline: no rotate moves, loads for
// one fragment set overlap the 8 WMMAs on the other set.
// Shapes: B=32 S=128 H=E=1024 V=32000 L=2; M%32==0, N%64==0, K%64==0 everywhere.

#define Bc 32
#define Sc 128
#define Hc 1024
#define Ec 1024
#define Vc 32000
#define Lc 2
#define Gc (3 * Hc)      // 3072 gates
#define Mrows (Bc * Sc)  // 4096

typedef __attribute__((ext_vector_type(16))) __bf16 v16bf;
typedef __attribute__((ext_vector_type(8)))  __bf16 v8bf;
typedef __attribute__((ext_vector_type(8)))  float  v8f;

// ---------- helpers ----------

__device__ __forceinline__ unsigned short f2bf(float f) {
    // round-to-nearest-even f32 -> bf16 bits
    unsigned int u = __float_as_uint(f);
    unsigned int r = u + 0x7FFFu + ((u >> 16) & 1u);
    return (unsigned short)(r >> 16);
}

__device__ __forceinline__ v16bf load_frag16(const __bf16* p0, const __bf16* p1) {
    // two 16-byte global loads -> one 16-element bf16 fragment
    v8bf lo = *reinterpret_cast<const v8bf*>(p0);
    v8bf hi = *reinterpret_cast<const v8bf*>(p1);
    v16bf r;
#pragma unroll
    for (int i = 0; i < 8; ++i) { r[i] = lo[i]; r[i + 8] = hi[i]; }
    return r;
}

// A fragment pair (two 16-row M-subtiles) at K offset k0.
// 16-bit A-frag layout: elem i -> K = (i<8?0:16) + kh*8 + (i&7)
__device__ __forceinline__ void load_a2(v16bf a[2], const __bf16* Arow0,
                                        const __bf16* Arow1, int k0, int kh) {
    a[0] = load_frag16(Arow0 + k0 + kh * 8, Arow0 + k0 + 16 + kh * 8);
    a[1] = load_frag16(Arow1 + k0 + kh * 8, Arow1 + k0 + 16 + kh * 8);
}

// B fragments for 4 N-subtiles at K offset k0.
// B-frag layout: elem i -> K = (i<8?0:16) + kh*8 + (i&7) (2x b128 loads)
__device__ __forceinline__ void load_b4(v16bf b[4], const __bf16* Wrow,
                                        size_t wsub, int k0, int kh) {
#pragma unroll
    for (int s = 0; s < 4; ++s) {
        const __bf16* bp = Wrow + (size_t)s * wsub + k0 + kh * 8;
        b[s] = load_frag16(bp, bp + 16);
    }
}

// 8 independent WMMAs: 2 M-subtiles x 4 N-subtiles
__device__ __forceinline__ void wmma8(v8f acc[2][4], const v16bf a[2],
                                      const v16bf b[4]) {
#pragma unroll
    for (int mi = 0; mi < 2; ++mi)
#pragma unroll
        for (int s = 0; s < 4; ++s)
            acc[mi][s] = __builtin_amdgcn_wmma_f32_16x16x32_bf16(
                false, a[mi], false, b[s], (short)0, acc[mi][s], false, false);
}

// ---------- cast / gather kernels ----------

__global__ void cast_f32_bf16_kernel(const float* __restrict__ src,
                                     unsigned short* __restrict__ dst, int n) {
    int i = blockIdx.x * blockDim.x + threadIdx.x;
    if (i < n) dst[i] = f2bf(src[i]);
}

__global__ void embed_gather_kernel(const int* __restrict__ tok,
                                    const float* __restrict__ table,
                                    unsigned short* __restrict__ dst) {
    int i = blockIdx.x * blockDim.x + threadIdx.x;  // over Mrows*Ec
    int row = i / Ec, e = i - row * Ec;
    int t = tok[row];
    dst[i] = f2bf(table[(size_t)t * Ec + e]);
}

__global__ void init_h_kernel(const float* __restrict__ h0,
                              float* __restrict__ hf,
                              unsigned short* __restrict__ hb) {
    int i = blockIdx.x * blockDim.x + threadIdx.x;  // over Bc*Hc
    float v = h0[i];
    hf[i] = v;
    hb[i] = f2bf(v);
}

// ---------- WMMA GEMM: C[M,N] = A[M,K](bf16) * W[N,K](bf16)^T + bias[N] ----------
// Requires M%32==0, N%64==0, K%64==0 (true for all call sites; K==1024).

template <bool NT>
__global__ void wmma_gemm_bf16_kernel(const unsigned short* __restrict__ Au,
                                      const unsigned short* __restrict__ Wu,
                                      const float* __restrict__ bias,
                                      float* __restrict__ C,
                                      int M, int N, int K) {
    const __bf16* A = reinterpret_cast<const __bf16*>(Au);
    const __bf16* W = reinterpret_cast<const __bf16*>(Wu);

    int wave = blockIdx.x * (blockDim.x >> 5) + (threadIdx.x >> 5);
    int lane = threadIdx.x & 31;
    int tiles_n = N >> 6;
    int m_tile = wave / tiles_n;
    int n_tile = wave - m_tile * tiles_n;
    if (m_tile * 32 >= M) return;

    int lr = lane & 15;   // row (A) / col (B,C) within 16-wide subtile
    int kh = lane >> 4;   // which K-half this lane holds

    const __bf16* Arow0 = A + (size_t)(m_tile * 32 + lr) * K;  // M-subtile 0
    const __bf16* Arow1 = Arow0 + (size_t)16 * K;              // M-subtile 1
    const __bf16* Wrow  = W + (size_t)(n_tile * 64 + lr) * K;
    const size_t wsub = (size_t)16 * K;

    v8f acc[2][4] = {};
    v16bf aP[2], bP[4], aQ[2], bQ[4];  // ping-pong fragment sets

    // prologue: fill set P with k=0
    load_a2(aP, Arow0, Arow1, 0, kh);
    load_b4(bP, Wrow, wsub, 0, kh);

    int k0 = 0;
    for (; k0 + 64 < K; k0 += 64) {
        // load Q(k0+32) while computing on P(k0)
        load_a2(aQ, Arow0, Arow1, k0 + 32, kh);
        load_b4(bQ, Wrow, wsub, k0 + 32, kh);
        wmma8(acc, aP, bP);
        // load P(k0+64) while computing on Q(k0+32)
        load_a2(aP, Arow0, Arow1, k0 + 64, kh);
        load_b4(bP, Wrow, wsub, k0 + 64, kh);
        wmma8(acc, aQ, bQ);
    }
    // tail: k0 == K-64; P holds K-64, load Q(K-32), drain both
    load_a2(aQ, Arow0, Arow1, K - 32, kh);
    load_b4(bQ, Wrow, wsub, K - 32, kh);
    wmma8(acc, aP, bP);
    wmma8(acc, aQ, bQ);

    // store: lane lr = col; VGPR r = row kh*8+r within each 16x16 subtile
    int nbase = n_tile * 64 + lr;
#pragma unroll
    for (int mi = 0; mi < 2; ++mi) {
        int mbase = m_tile * 32 + mi * 16 + kh * 8;
#pragma unroll
        for (int s = 0; s < 4; ++s) {
            int n = nbase + s * 16;
            float bv = bias[n];
            float* Cp = C + (size_t)mbase * N + n;
#pragma unroll
            for (int r = 0; r < 8; ++r) {
                float v = acc[mi][s][r] + bv;
                if (NT)
                    __builtin_nontemporal_store(v, Cp + (size_t)r * N);
                else
                    Cp[(size_t)r * N] = v;
            }
        }
    }
}

// ---------- GRU gate update (PyTorch order r,z,n) ----------
// xW: [Mrows,3H] precomputed input projection; gh: [B,3H] = h@Whh^T + b_hh
__global__ void gru_gate_kernel(const float* __restrict__ xW,
                                const float* __restrict__ gh,
                                float* __restrict__ hf,
                                unsigned short* __restrict__ hb,
                                unsigned short* __restrict__ xout,
                                int t) {
    int i = blockIdx.x * blockDim.x + threadIdx.x;  // over Bc*Hc
    int b = i / Hc, j = i - b * Hc;
    size_t row = (size_t)(b * Sc + t);
    const float* xwr = xW + row * Gc;
    const float* ghr = gh + (size_t)b * Gc;

    float xr = xwr[j], xz = xwr[Hc + j], xn = xwr[2 * Hc + j];
    float hr = ghr[j], hz = ghr[Hc + j], hn = ghr[2 * Hc + j];

    float r = 1.0f / (1.0f + __expf(-(xr + hr)));
    float z = 1.0f / (1.0f + __expf(-(xz + hz)));
    float n = tanhf(xn + r * hn);
    float hprev = hf[i];
    float hnew = (1.0f - z) * n + z * hprev;

    hf[i] = hnew;
    hb[i] = f2bf(hnew);
    xout[row * Hc + j] = f2bf(hnew);
}

// ---------- host side ----------

extern "C" void kernel_launch(void* const* d_in, const int* in_sizes, int n_in,
                              void* d_out, int out_size, void* d_ws, size_t ws_size,
                              hipStream_t stream) {
    const int*   inputs = (const int*)  d_in[0];   // [B,S]
    // d_in[1] = lengths: unused by the reference computation
    const float* hidden = (const float*)d_in[2];   // [L,B,H]
    const float* embed  = (const float*)d_in[3];   // [V,E]
    const float* W_ih   = (const float*)d_in[4];   // [L,3H,E]
    const float* W_hh   = (const float*)d_in[5];   // [L,3H,H]
    const float* b_ih   = (const float*)d_in[6];   // [L,3H]
    const float* b_hh   = (const float*)d_in[7];   // [L,3H]
    const float* W_out  = (const float*)d_in[8];   // [V,H]
    const float* b_out  = (const float*)d_in[9];   // [V]
    float* logits = (float*)d_out;                 // [B,S,V]

    // workspace carve-up (256B aligned); total ~159 MB
    char* ws = (char*)d_ws;
    size_t off = 0;
    auto take = [&](size_t bytes) -> char* {
        char* p = ws + off;
        off = (off + bytes + 255) & ~(size_t)255;
        return p;
    };
    unsigned short* wih_h  = (unsigned short*)take((size_t)Lc * Gc * Ec * 2);
    unsigned short* whh_h  = (unsigned short*)take((size_t)Lc * Gc * Hc * 2);
    unsigned short* wout_h = (unsigned short*)take((size_t)Vc * Hc * 2);
    unsigned short* xa     = (unsigned short*)take((size_t)Mrows * Ec * 2);
    unsigned short* xb     = (unsigned short*)take((size_t)Mrows * Hc * 2);
    float*          xw     = (float*)         take((size_t)Mrows * Gc * 4);
    float*          ghbuf  = (float*)         take((size_t)Bc * Gc * 4);
    float*          hf     = (float*)         take((size_t)Lc * Bc * Hc * 4);
    unsigned short* hb     = (unsigned short*)take((size_t)Lc * Bc * Hc * 2);
    (void)ws_size; (void)in_sizes; (void)n_in; (void)out_size;

    auto cast = [&](const float* s, unsigned short* d, int n) {
        cast_f32_bf16_kernel<<<(n + 255) / 256, 256, 0, stream>>>(s, d, n);
    };
    auto gemm = [&](const unsigned short* A, const unsigned short* Wt,
                    const float* bias, float* C, int M, int N, int K, bool nt) {
        int waves  = (M / 32) * (N / 64);
        int blocks = (waves + 3) / 4;  // 4 waves (128 threads) per block
        if (nt)
            wmma_gemm_bf16_kernel<true><<<blocks, 128, 0, stream>>>(A, Wt, bias, C, M, N, K);
        else
            wmma_gemm_bf16_kernel<false><<<blocks, 128, 0, stream>>>(A, Wt, bias, C, M, N, K);
    };

    // 1. weights -> bf16
    cast(W_ih,  wih_h,  Lc * Gc * Ec);
    cast(W_hh,  whh_h,  Lc * Gc * Hc);
    cast(W_out, wout_h, Vc * Hc);

    // 2. embedding gather -> bf16 activations [Mrows, E]
    embed_gather_kernel<<<(Mrows * Ec) / 256, 256, 0, stream>>>(inputs, embed, xa);

    // 3. two GRU layers
    unsigned short* xin = xa;
    unsigned short* xout = xb;
    for (int l = 0; l < Lc; ++l) {
        float*          hfl = hf + (size_t)l * Bc * Hc;
        unsigned short* hbl = hb + (size_t)l * Bc * Hc;
        init_h_kernel<<<(Bc * Hc) / 256, 256, 0, stream>>>(
            hidden + (size_t)l * Bc * Hc, hfl, hbl);

        // input projection for the whole sequence: xw = x @ W_ih^T + b_ih
        // (xw is re-read 128x by gate kernels -> regular RT stores)
        gemm(xin, wih_h + (size_t)l * Gc * Ec, b_ih + (size_t)l * Gc,
             xw, Mrows, Gc, Ec, false);

        for (int t = 0; t < Sc; ++t) {
            // gh = h @ W_hh^T + b_hh   (M=32 -> 1 wave M-tile)
            gemm(hbl, whh_h + (size_t)l * Gc * Hc, b_hh + (size_t)l * Gc,
                 ghbuf, Bc, Gc, Hc, false);
            gru_gate_kernel<<<(Bc * Hc) / 256, 256, 0, stream>>>(
                xw, ghbuf, hfl, hbl, xout, t);
        }
        // layer output becomes next layer's input
        unsigned short* tmp = xin; xin = xout; xout = tmp;
    }

    // 4. vocab projection: logits = x2 @ W_out^T + b_out (268 GFLOP).
    //    Output is write-once 524 MB -> nontemporal stores keep W_out hot in L2.
    gemm(xin, wout_h, b_out, logits, Mrows, Vc, Hc, true);
}
